// RNNTextGenerativeModel_52235392254613
// MI455X (gfx1250) — compile-verified, matrix operand order checked by reading, not run on hardware
//
#include <hip/hip_runtime.h>
#include <hip/hip_bf16.h>

// ---------------------------------------------------------------------------
// Types for WMMA (CDNA5 gfx1250, wave32)
// ---------------------------------------------------------------------------
typedef __attribute__((ext_vector_type(16))) __bf16    v16bf;
typedef __attribute__((ext_vector_type(8)))  float     v8f;
typedef __attribute__((ext_vector_type(4)))  unsigned  v4u;

#define DIMZ   128
#define VOCAB  32000
#define HID    1024
#define EMB    512
#define SOS    1
#define MAXLEN 128
#define BATCH  32

// ---------------------------------------------------------------------------
// Helpers
// ---------------------------------------------------------------------------
__device__ __forceinline__ unsigned short f32_to_bf16(float f) {
  unsigned u = __float_as_uint(f);
  unsigned rnd = 0x7FFFu + ((u >> 16) & 1u);   // round-to-nearest-even
  return (unsigned short)((u + rnd) >> 16);
}

// Load one wave's 16x32 bf16 A (or 32x16 B, stored N-major) fragment.
// ISA 16-bit A layout: lanes 0-15 hold row M=lane, K-chunks {k0..k0+7, k0+16..k0+23};
// lanes 16-31 hold the {k0+8..k0+15, k0+24..k0+31} halves.  Two b128 loads/lane.
__device__ __forceinline__ v16bf load_frag16(const unsigned short* __restrict__ base,
                                             int ld, int row, int k0, int lane) {
  int klo = k0 + ((lane & 16) ? 8 : 0);
  const unsigned short* p = base + (size_t)row * ld + klo;
  union { v16bf v; v4u q[2]; } f;
  f.q[0] = *(const v4u*)(p);
  f.q[1] = *(const v4u*)(p + 16);
  return f.v;
}

// ---------------------------------------------------------------------------
// f32 -> bf16 weight conversion (once per call; makes W_out L2-resident @65MB)
// ---------------------------------------------------------------------------
__global__ void cvt_bf16_kernel(const float* __restrict__ in,
                                unsigned short* __restrict__ out, int n) {
  int i = blockIdx.x * blockDim.x + threadIdx.x;
  if (i < n) out[i] = f32_to_bf16(in[i]);
}

// ---------------------------------------------------------------------------
// h0 = z @ W_lh^T + b_lh ;  x = SOS
// ---------------------------------------------------------------------------
__global__ void h0_init_kernel(const float* __restrict__ z,
                               const float* __restrict__ W_lh,
                               const float* __restrict__ b_lh,
                               float* __restrict__ h,
                               unsigned short* __restrict__ h_bf,
                               int* __restrict__ x) {
  int idx = blockIdx.x * blockDim.x + threadIdx.x;
  if (idx < BATCH) x[idx] = SOS;
  if (idx >= BATCH * HID) return;
  int b = idx >> 10, j = idx & (HID - 1);
  const float* zr = z + b * DIMZ;
  const float* wr = W_lh + (size_t)j * DIMZ;
  float acc = b_lh[j];
  #pragma unroll 4
  for (int k = 0; k < DIMZ; ++k) acc += zr[k] * wr[k];
  h[idx]    = acc;
  h_bf[idx] = f32_to_bf16(acc);
}

// ---------------------------------------------------------------------------
// e = bf16(emb[x])   [32,512]
// ---------------------------------------------------------------------------
__global__ void embed_kernel(const float* __restrict__ emb,
                             const int* __restrict__ x,
                             unsigned short* __restrict__ e_bf) {
  int idx = blockIdx.x * blockDim.x + threadIdx.x;
  if (idx >= BATCH * EMB) return;
  int b = idx >> 9, k = idx & (EMB - 1);
  e_bf[idx] = f32_to_bf16(emb[(size_t)x[b] * EMB + k]);
}

// ---------------------------------------------------------------------------
// Generic WMMA GEMM:  out[M=32, N] = A[32,K](bf16) * B[N,K](bf16)^T + bias[N]
// 64 threads = 2 waves.  Wave w owns m-tile m0=16w and BOTH n-tiles
// {32*blockIdx.x, 32*blockIdx.x+16}: A fragment loaded once, used by two
// v_wmma issues per k-step.  The k-loop is software-pipelined (double
// buffered) so next-chunk loads are in flight across the current WMMAs.
// ---------------------------------------------------------------------------
__global__ __launch_bounds__(64)
void wmma_gemm_bias_kernel(const unsigned short* __restrict__ A,
                           const unsigned short* __restrict__ B,
                           const float* __restrict__ bias,
                           float* __restrict__ out,
                           int K, int ldout) {
  int lane  = threadIdx.x & 31;
  int m0    = (threadIdx.x >> 5) << 4;
  int n0    = blockIdx.x << 5;
  int arow  = m0 + (lane & 15);
  int brow0 = n0 + (lane & 15);
  int brow1 = brow0 + 16;
  const unsigned short* bptr0 = B + (size_t)brow0 * K;
  const unsigned short* bptr1 = B + (size_t)brow1 * K;

  v8f acc0 = {};
  v8f acc1 = {};

  // prologue: first k-chunk in flight
  v16bf a  = load_frag16(A, K, arow,  0, lane);
  v16bf b0 = load_frag16(B, K, brow0, 0, lane);
  v16bf b1 = load_frag16(B, K, brow1, 0, lane);

  for (int k0 = 0; k0 < K; k0 += 32) {
    // issue next chunk's loads BEFORE consuming the current fragments so the
    // compiler can wait with slack (s_wait_loadcnt > 0) and overlap latency.
    int kn = (k0 + 32 < K) ? (k0 + 32) : k0;       // clamp: last iter reloads (harmless)
    if (k0 + 512 < K) {
      __builtin_prefetch(bptr0 + k0 + 512, 0, 1);  // global_prefetch_b8
      __builtin_prefetch(bptr1 + k0 + 512, 0, 1);
    }
    v16bf an  = load_frag16(A, K, arow,  kn, lane);
    v16bf b0n = load_frag16(B, K, brow0, kn, lane);
    v16bf b1n = load_frag16(B, K, brow1, kn, lane);

    acc0 = __builtin_amdgcn_wmma_f32_16x16x32_bf16(
        false, a, false, b0, (short)0, acc0, false, false);
    acc1 = __builtin_amdgcn_wmma_f32_16x16x32_bf16(
        false, a, false, b1, (short)0, acc1, false, false);

    a = an; b0 = b0n; b1 = b1n;
  }

  // D layout: VGPR i -> M = m0 + i (lanes 0-15) or m0 + 8 + i (lanes 16-31); N = n-tile + (lane&15)
  int nA = n0 + (lane & 15);
  int nB = nA + 16;
  float bvA = bias[nA];
  float bvB = bias[nB];
  int mbase = m0 + ((lane & 16) ? 8 : 0);
  #pragma unroll
  for (int i = 0; i < 8; ++i) {
    out[(size_t)(mbase + i) * ldout + nA] = acc0[i] + bvA;
    out[(size_t)(mbase + i) * ldout + nB] = acc1[i] + bvB;
  }
}

// ---------------------------------------------------------------------------
// GRU pointwise update:  h = (1-z)*n + z*h  (torch gate order r,z,n)
// ---------------------------------------------------------------------------
__global__ void gru_update_kernel(const float* __restrict__ gi,
                                  const float* __restrict__ gh,
                                  float* __restrict__ h,
                                  unsigned short* __restrict__ h_bf) {
  int idx = blockIdx.x * blockDim.x + threadIdx.x;
  if (idx >= BATCH * HID) return;
  int b = idx >> 10, j = idx & (HID - 1);
  const float* gir = gi + (size_t)b * 3 * HID;
  const float* ghr = gh + (size_t)b * 3 * HID;
  float r  = 1.0f / (1.0f + expf(-(gir[j]           + ghr[j])));
  float zg = 1.0f / (1.0f + expf(-(gir[HID + j]     + ghr[HID + j])));
  float n  = tanhf(gir[2 * HID + j] + r * ghr[2 * HID + j]);
  float hn = (1.0f - zg) * n + zg * h[idx];
  h[idx]    = hn;
  h_bf[idx] = f32_to_bf16(hn);
}

// ---------------------------------------------------------------------------
// Per-batch-row log_softmax + greedy argmax; writes logp[b,t,:], sample[b,t], x_next[b]
// ---------------------------------------------------------------------------
__global__ void softmax_argmax_kernel(const float* __restrict__ logits,
                                      float* __restrict__ logp,
                                      float* __restrict__ sample,
                                      int* __restrict__ x_next,
                                      int t) {
  __shared__ float smax[256];
  __shared__ int   sidx[256];
  __shared__ float ssum[256];
  int b = blockIdx.x, tid = threadIdx.x;
  const float* row = logits + (size_t)b * VOCAB;

  float m = -__builtin_inff(); int mi = 0;
  for (int v = tid; v < VOCAB; v += 256) {
    float xv = row[v];
    if (xv > m) { m = xv; mi = v; }
  }
  smax[tid] = m; sidx[tid] = mi;
  __syncthreads();
  for (int s = 128; s > 0; s >>= 1) {
    if (tid < s) {
      if (smax[tid + s] > smax[tid] ||
          (smax[tid + s] == smax[tid] && sidx[tid + s] < sidx[tid])) {
        smax[tid] = smax[tid + s]; sidx[tid] = sidx[tid + s];
      }
    }
    __syncthreads();
  }
  float rowmax = smax[0];
  int   amax   = sidx[0];

  float sum = 0.0f;
  for (int v = tid; v < VOCAB; v += 256) sum += expf(row[v] - rowmax);
  ssum[tid] = sum;
  __syncthreads();
  for (int s = 128; s > 0; s >>= 1) {
    if (tid < s) ssum[tid] += ssum[tid + s];
    __syncthreads();
  }
  float logZ = rowmax + logf(ssum[0]);

  float* lp = logp + ((size_t)b * MAXLEN + t) * VOCAB;
  for (int v = tid; v < VOCAB; v += 256) lp[v] = row[v] - logZ;

  if (tid == 0) {
    x_next[b] = amax;
    sample[b * MAXLEN + t] = (float)amax;
  }
}

// ---------------------------------------------------------------------------
// Host launcher
// ---------------------------------------------------------------------------
extern "C" void kernel_launch(void* const* d_in, const int* in_sizes, int n_in,
                              void* d_out, int out_size, void* d_ws, size_t ws_size,
                              hipStream_t stream) {
  (void)in_sizes; (void)n_in; (void)out_size; (void)ws_size;
  const float* z     = (const float*)d_in[0];
  const float* emb   = (const float*)d_in[1];
  const float* W_lh  = (const float*)d_in[2];
  const float* b_lh  = (const float*)d_in[3];
  const float* w_ih  = (const float*)d_in[4];
  const float* w_hh  = (const float*)d_in[5];
  const float* b_ih  = (const float*)d_in[6];
  const float* b_hh  = (const float*)d_in[7];
  const float* W_out = (const float*)d_in[8];
  const float* b_out = (const float*)d_in[9];

  float* logp   = (float*)d_out;                                   // [32,128,32000]
  float* sample = logp + (size_t)BATCH * MAXLEN * VOCAB;           // [32,128]

  // Workspace carve-out (256B aligned)
  char* ws = (char*)d_ws;
  size_t off = 0;
  auto alloc = [&](size_t bytes) -> void* {
    void* p = ws + off;
    off += (bytes + 255) & ~(size_t)255;
    return p;
  };
  unsigned short* w_ih_bf  = (unsigned short*)alloc((size_t)3 * HID * EMB * 2);
  unsigned short* w_hh_bf  = (unsigned short*)alloc((size_t)3 * HID * HID * 2);
  unsigned short* W_out_bf = (unsigned short*)alloc((size_t)VOCAB * HID * 2);   // 64MB, L2-resident
  unsigned short* e_bf     = (unsigned short*)alloc((size_t)BATCH * EMB * 2);
  unsigned short* h_bf     = (unsigned short*)alloc((size_t)BATCH * HID * 2);
  float*          h_f32    = (float*)alloc((size_t)BATCH * HID * 4);
  float*          gi       = (float*)alloc((size_t)BATCH * 3 * HID * 4);
  float*          gh       = (float*)alloc((size_t)BATCH * 3 * HID * 4);
  float*          logits   = (float*)alloc((size_t)BATCH * VOCAB * 4);
  int*            x_cur    = (int*)alloc(BATCH * 4);

  // One-time (per call) bf16 weight conversion
  int n;
  n = 3 * HID * EMB;
  cvt_bf16_kernel<<<(n + 255) / 256, 256, 0, stream>>>(w_ih, w_ih_bf, n);
  n = 3 * HID * HID;
  cvt_bf16_kernel<<<(n + 255) / 256, 256, 0, stream>>>(w_hh, w_hh_bf, n);
  n = VOCAB * HID;
  cvt_bf16_kernel<<<(n + 255) / 256, 256, 0, stream>>>(W_out, W_out_bf, n);

  // h0 = z @ W_lh^T + b_lh ; x = SOS
  h0_init_kernel<<<(BATCH * HID + 255) / 256, 256, 0, stream>>>(
      z, W_lh, b_lh, h_f32, h_bf, x_cur);

  for (int t = 0; t < MAXLEN; ++t) {
    embed_kernel<<<(BATCH * EMB + 255) / 256, 256, 0, stream>>>(emb, x_cur, e_bf);
    // gi = e @ w_ih^T + b_ih   (N=3072, K=512) : 96 blocks of 32x32 tiles
    wmma_gemm_bias_kernel<<<3 * HID / 32, 64, 0, stream>>>(e_bf, w_ih_bf, b_ih, gi, EMB, 3 * HID);
    // gh = h @ w_hh^T + b_hh   (N=3072, K=1024)
    wmma_gemm_bias_kernel<<<3 * HID / 32, 64, 0, stream>>>(h_bf, w_hh_bf, b_hh, gh, HID, 3 * HID);
    gru_update_kernel<<<(BATCH * HID + 255) / 256, 256, 0, stream>>>(gi, gh, h_f32, h_bf);
    // logits = h @ W_out^T + b_out   (N=32000, K=1024) : 1000 blocks
    wmma_gemm_bias_kernel<<<VOCAB / 32, 64, 0, stream>>>(h_bf, W_out_bf, b_out, logits, HID, VOCAB);
    softmax_argmax_kernel<<<BATCH, 256, 0, stream>>>(logits, logp, sample, x_cur, t);
  }
}
